// SpatialSelfAttentionLayer_39711267618988
// MI455X (gfx1250) — matrix operand order repaired and checked by reading, not compiled, not run on hardware
//
#include <hip/hip_runtime.h>
#include <hip/hip_bf16.h>
#include <cstdint>
#include <cstddef>

typedef __attribute__((ext_vector_type(16))) _Float16 v16h;
typedef __attribute__((ext_vector_type(8)))  float    v8f;

namespace {

constexpr int Bc = 8, Lc = 1024, Dc = 256, Hc = 8, DHc = 32;
constexpr float kScale = 0.17677669529663687f; // 1/sqrt(32)
constexpr int kRowStride = 324;                // LDS row stride (floats): 16B-aligned,
                                               // ==4 mod 64 banks -> conflict-friendly

union ABu { v16h v; uint4 q[2]; unsigned int u[8]; };
union PKU { unsigned int u; _Float16 h[2]; };

__device__ __forceinline__ v8f wmma16(v16h a, v16h b, v8f c) {
  // D = A(16x32 f16) * B(32x16 f16) + C(16x16 f32)
  return __builtin_amdgcn_wmma_f32_16x16x32_f16(false, a, false, b, (short)0, c,
                                                false, false);
}

// xor-16 lane swap within the wave32 (ds_swizzle SWAPX16: xor=0x10, and=0x1f)
__device__ __forceinline__ float swapx16f(float x) {
  return __int_as_float(__builtin_amdgcn_ds_swizzle(__float_as_int(x), 0x401f));
}
__device__ __forceinline__ unsigned int swapx16u(unsigned int x) {
  return (unsigned int)__builtin_amdgcn_ds_swizzle((int)x, 0x401f);
}

// CDNA5 async global->LDS copy (16B per lane), tracked by ASYNCcnt
__device__ __forceinline__ void async_g2lds_b128(unsigned ldsoff, const void* gaddr) {
  asm volatile("global_load_async_to_lds_b128 %0, %1, off"
               :: "v"(ldsoff), "v"((unsigned long long)gaddr)
               : "memory");
}
__device__ __forceinline__ void wait_asynccnt0() {
  asm volatile("s_wait_asynccnt 0" ::: "memory");
}

} // namespace

// ---------------------------------------------------------------------------
// prep: tgt->f16, qkin=(tgt+qpos)->f16, weights transposed+converted to f16
// ---------------------------------------------------------------------------
__global__ void prep_kernel(const float* __restrict__ tgt, const float* __restrict__ qpos,
                            const float* __restrict__ Wq, const float* __restrict__ Wk,
                            const float* __restrict__ Wv, const float* __restrict__ Wo,
                            _Float16* __restrict__ tgtf, _Float16* __restrict__ qkinf,
                            _Float16* __restrict__ Wt) {
  int i = blockIdx.x * blockDim.x + threadIdx.x;
  if (i < Bc * Lc * Dc) {
    float t = tgt[i];
    tgtf[i]  = (_Float16)t;
    qkinf[i] = (_Float16)(t + qpos[i]);
  }
  if (i < 4 * Dc * Dc) {  // Wt[w][n][k] = W[w][k][n]
    int w = i >> 16, rem = i & 65535;
    int n = rem >> 8, k = rem & 255;
    const float* W = (w == 0) ? Wq : (w == 1) ? Wk : (w == 2) ? Wv : Wo;
    Wt[i] = (_Float16)W[k * Dc + n];
  }
}

// ---------------------------------------------------------------------------
// proj: out = X(8192x256 f16) @ W^T-stored(256x256 f16) + bias, WMMA tiles.
// transposed==0 -> row-major f16 [B,L,D]; transposed==1 -> V^T [B,H,DH,L]
// ---------------------------------------------------------------------------
__global__ void __launch_bounds__(256)
proj_kernel(const _Float16* __restrict__ X, const _Float16* __restrict__ Wt,
            const float* __restrict__ bias,
            _Float16* __restrict__ outRM, _Float16* __restrict__ outTR,
            int transposed) {
  int wave = (blockIdx.x * blockDim.x + threadIdx.x) >> 5;
  int lane = threadIdx.x & 31;
  int rt = wave >> 4, ct = wave & 15;
  int m0 = rt << 4, n0 = ct << 4;
  int lq = lane & 15, hi = lane >> 4;

  v8f acc = {0.f, 0.f, 0.f, 0.f, 0.f, 0.f, 0.f, 0.f};
  const _Float16* arow = X + (size_t)(m0 + lq) * Dc;
  const _Float16* brow = Wt + (size_t)(n0 + lq) * Dc;
#pragma unroll
  for (int k0 = 0; k0 < Dc; k0 += 32) {
    ABu a, b;
    a.q[0] = *reinterpret_cast<const uint4*>(arow + k0 + hi * 8);
    a.q[1] = *reinterpret_cast<const uint4*>(arow + k0 + 16 + hi * 8);
    b.q[0] = *reinterpret_cast<const uint4*>(brow + k0 + hi * 8);
    b.q[1] = *reinterpret_cast<const uint4*>(brow + k0 + 16 + hi * 8);
    acc = wmma16(a.v, b.v, acc);
  }
  float bn = bias[n0 + lq];
  if (!transposed) {
#pragma unroll
    for (int r = 0; r < 8; ++r) {
      int g = m0 + r + 8 * hi;  // D-layout: row = r + 8*(lane/16), col = lane%16
      outRM[(size_t)g * Dc + n0 + lq] = (_Float16)(acc[r] + bn);
    }
  } else {
    int dg = n0 + lq, hh = dg >> 5, dh = dg & 31;
#pragma unroll
    for (int r = 0; r < 8; ++r) {
      int g = m0 + r + 8 * hi;
      int bb = g >> 10, l = g & (Lc - 1);
      outTR[((size_t)(bb * Hc + hh) * DHc + dh) * Lc + l] = (_Float16)(acc[r] + bn);
    }
  }
}

// ---------------------------------------------------------------------------
// attn: one 256-thread block = 8 waves = 8 heads sharing one (b, 16-query tile).
// pairwise_locs is head-independent: staged into LDS ONCE per block via
// global_load_async_to_lds_b128 (ASYNCcnt). softmax(log(clip(loc))+attn) is
// computed as clip(loc)*exp(attn - m) -- no logs, half the transcendentals.
// ---------------------------------------------------------------------------
__global__ void __launch_bounds__(256)
attn_kernel(const _Float16* __restrict__ Qf, const _Float16* __restrict__ Kf,
            const _Float16* __restrict__ Vt, const float* __restrict__ plocs,
            const unsigned char* __restrict__ kpm,
            const float* __restrict__ Wl, const float* __restrict__ blv,
            _Float16* __restrict__ att) {
  __shared__ float sloc[16 * kRowStride];  // 20.25 KB staged pairwise chunk

  int tid  = threadIdx.x;
  int h    = tid >> 5;       // wave index == head
  int lane = tid & 31;
  int b  = blockIdx.x >> 6;
  int q0 = (blockIdx.x & 63) << 4;
  int lq = lane & 15, hi = lane >> 4;

  // Q^T as B-operand (fixed over chunk loop): lane n holds Q[q0+n][dims 8*hi.., 16+8*hi..]
  ABu qb;
  const _Float16* qptr = Qf + (size_t)(b * Lc + q0 + lq) * Dc + h * DHc + hi * 8;
  qb.q[0] = *reinterpret_cast<const uint4*>(qptr);
  qb.q[1] = *reinterpret_cast<const uint4*>(qptr + 16);

  float wl0 = Wl[0 * Hc + h], wl1 = Wl[1 * Hc + h], wl2 = Wl[2 * Hc + h],
        wl3 = Wl[3 * Hc + h], wl4 = Wl[4 * Hc + h];
  float blh = blv[h];

  float m_i = -1e30f, l_i = 0.f;
  v8f o0 = {0.f, 0.f, 0.f, 0.f, 0.f, 0.f, 0.f, 0.f};
  v8f o1 = {0.f, 0.f, 0.f, 0.f, 0.f, 0.f, 0.f, 0.f};
  const v8f z8 = {0.f, 0.f, 0.f, 0.f, 0.f, 0.f, 0.f, 0.f};

  const unsigned char* kpb = kpm + b * Lc;
  // base of the 16 pairwise rows for this query tile (floats)
  const float* gsrc0 = plocs + ((size_t)(b * Lc + q0) * Lc) * 5;

  // precompute this thread's staging slots (5 float4 per thread, uniform)
  unsigned ldsoff_s[5];
  const float* gp_s[5];
  {
    const float* gsrc = gsrc0;
#pragma unroll
    for (int k = 0; k < 5; ++k) {
      int idx = k * 256 + tid;          // float4 index, 0..1279
      int row = idx / 80, col = idx - row * 80;
      gp_s[k] = gsrc + (size_t)row * (Lc * 5) + col * 4;
      ldsoff_s[k] = (unsigned)(uintptr_t)(sloc + row * kRowStride + col * 4);
    }
  }

  for (int t0 = 0; t0 < Lc; t0 += 64) {
    // ---- async stage: 16 rows x 64 keys x 5 f32 = 1280 float4 -> LDS ----
#pragma unroll
    for (int k = 0; k < 5; ++k) {
      const float* gp = gp_s[k] + (size_t)t0 * 5;
      async_g2lds_b128(ldsoff_s[k], gp);
      if (t0 + 64 < Lc)  // prefetch next stage into L2 (global_prefetch_b8)
        __builtin_prefetch(gp + 320, 0, 1);
    }
    // per-stage padding-mask bytes for this lane's key slots (4x u64)
    uint64_t mk0 = *reinterpret_cast<const uint64_t*>(kpb + t0 + 8 * hi);
    uint64_t mk1 = *reinterpret_cast<const uint64_t*>(kpb + t0 + 16 + 8 * hi);
    uint64_t mk2 = *reinterpret_cast<const uint64_t*>(kpb + t0 + 32 + 8 * hi);
    uint64_t mk3 = *reinterpret_cast<const uint64_t*>(kpb + t0 + 48 + 8 * hi);
    wait_asynccnt0();
    __syncthreads();

#pragma unroll
    for (int s = 0; s < 2; ++s) {
      int tt0 = t0 + s * 32;
      uint64_t mka = s ? mk2 : mk0;
      uint64_t mkb = s ? mk3 : mk1;

      // K chunk as two A-operand tiles (rows = keys)
      ABu ka, kb;
      const _Float16* kpA = Kf + (size_t)(b * Lc + tt0 + lq) * Dc + h * DHc;
      ka.q[0] = *reinterpret_cast<const uint4*>(kpA + hi * 8);
      ka.q[1] = *reinterpret_cast<const uint4*>(kpA + 16 + hi * 8);
      const _Float16* kpB = kpA + 16 * Dc;
      kb.q[0] = *reinterpret_cast<const uint4*>(kpB + hi * 8);
      kb.q[1] = *reinterpret_cast<const uint4*>(kpB + 16 + hi * 8);

      v8f sA = wmma16(ka.v, qb.v, z8);  // S^T keys tt0..tt0+15
      v8f sB = wmma16(kb.v, qb.v, z8);  // S^T keys tt0+16..tt0+31

      // attn logits + clipped loc factor (relu folded into the 1e-6 floor)
      const float* lrow = sloc + lq * kRowStride;
      float la[8], lb[8], ca[8], cb[8];
#pragma unroll
      for (int r = 0; r < 8; ++r) {
        int tla = s * 32 + 8 * hi + r;  // key offset within 64-key stage
        int tlb = tla + 16;
        const float* pa = lrow + tla * 5;
        float va = fmaf(pa[0], wl0, fmaf(pa[1], wl1, fmaf(pa[2], wl2,
                   fmaf(pa[3], wl3, fmaf(pa[4], wl4, blh)))));
        ca[r] = fmaxf(va, 1e-6f);
        float ga = sA[r] * kScale;
        if ((mka >> (8 * r)) & 0xffull) ga = -1e30f;
        la[r] = ga;
        const float* pb = lrow + tlb * 5;
        float vb = fmaf(pb[0], wl0, fmaf(pb[1], wl1, fmaf(pb[2], wl2,
                   fmaf(pb[3], wl3, fmaf(pb[4], wl4, blh)))));
        cb[r] = fmaxf(vb, 1e-6f);
        float gb = sB[r] * kScale;
        if ((mkb >> (8 * r)) & 0xffull) gb = -1e30f;
        lb[r] = gb;
      }

      // online softmax on the attn part; loc factor applied multiplicatively
      float mc = -1e30f;
#pragma unroll
      for (int r = 0; r < 8; ++r) mc = fmaxf(mc, fmaxf(la[r], lb[r]));
      mc = fmaxf(mc, swapx16f(mc));
      float m_new = fmaxf(m_i, mc);
      float alpha = __expf(m_i - m_new);
      float ssum = 0.f;
#pragma unroll
      for (int r = 0; r < 8; ++r) {
        ca[r] = ca[r] * __expf(la[r] - m_new);   // p = clip(loc)*exp(attn-m)
        cb[r] = cb[r] * __expf(lb[r] - m_new);
        ssum += ca[r] + cb[r];
      }
      ssum += swapx16f(ssum);
      l_i = l_i * alpha + ssum;
      m_i = m_new;
#pragma unroll
      for (int r = 0; r < 8; ++r) { o0[r] *= alpha; o1[r] *= alpha; }

      // pack P to f16 pairs and relayout D->B-operand with xor16 swizzles
      unsigned int pa4[4], pb4[4];
#pragma unroll
      for (int v = 0; v < 4; ++v) {
        PKU t1; t1.h[0] = (_Float16)ca[2 * v]; t1.h[1] = (_Float16)ca[2 * v + 1];
        pa4[v] = t1.u;
        PKU t2; t2.h[0] = (_Float16)cb[2 * v]; t2.h[1] = (_Float16)cb[2 * v + 1];
        pb4[v] = t2.u;
      }
      ABu bp;
#pragma unroll
      for (int v = 0; v < 4; ++v) {
        unsigned int spa = swapx16u(pa4[v]);
        unsigned int spb = swapx16u(pb4[v]);
        bp.u[v]     = hi ? spb : pa4[v];  // keys 16*hi+0..7  (this lane's column)
        bp.u[4 + v] = hi ? pb4[v] : spa;  // keys 16*hi+8..15
      }

      // V^T chunk as two A-operand tiles (rows = dims, contiguous along tokens)
      ABu v0, v1;
      const _Float16* vp0 =
          Vt + ((size_t)(b * Hc + h) * DHc + lq) * Lc + tt0 + hi * 16;
      v0.q[0] = *reinterpret_cast<const uint4*>(vp0);
      v0.q[1] = *reinterpret_cast<const uint4*>(vp0 + 8);
      const _Float16* vp1 = vp0 + 16 * Lc;
      v1.q[0] = *reinterpret_cast<const uint4*>(vp1);
      v1.q[1] = *reinterpret_cast<const uint4*>(vp1 + 8);

      o0 = wmma16(v0.v, bp.v, o0);  // O^T dims 0..15
      o1 = wmma16(v1.v, bp.v, o1);  // O^T dims 16..31
    }
    __syncthreads();
  }

  float inv = 1.f / l_i;
  _Float16* ob = att + (size_t)(b * Lc + q0 + lq) * Dc + h * DHc;
#pragma unroll
  for (int r = 0; r < 8; r += 2) {
    PKU t1; t1.h[0] = (_Float16)(o0[r] * inv); t1.h[1] = (_Float16)(o0[r + 1] * inv);
    *reinterpret_cast<unsigned int*>(ob + 8 * hi + r) = t1.u;
    PKU t2; t2.h[0] = (_Float16)(o1[r] * inv); t2.h[1] = (_Float16)(o1[r + 1] * inv);
    *reinterpret_cast<unsigned int*>(ob + 16 + 8 * hi + r) = t2.u;
  }
}

// ---------------------------------------------------------------------------
// oproj: xres = tgt + att @ Wo + bo  (f32)
// ---------------------------------------------------------------------------
__global__ void __launch_bounds__(256)
oproj_kernel(const _Float16* __restrict__ att, const _Float16* __restrict__ WoT,
             const float* __restrict__ bo, const float* __restrict__ tgt,
             float* __restrict__ xres) {
  int wave = (blockIdx.x * blockDim.x + threadIdx.x) >> 5;
  int lane = threadIdx.x & 31;
  int rt = wave >> 4, ct = wave & 15;
  int m0 = rt << 4, n0 = ct << 4;
  int lq = lane & 15, hi = lane >> 4;

  v8f acc = {0.f, 0.f, 0.f, 0.f, 0.f, 0.f, 0.f, 0.f};
  const _Float16* arow = att + (size_t)(m0 + lq) * Dc;
  const _Float16* brow = WoT + (size_t)(n0 + lq) * Dc;
#pragma unroll
  for (int k0 = 0; k0 < Dc; k0 += 32) {
    ABu a, b;
    a.q[0] = *reinterpret_cast<const uint4*>(arow + k0 + hi * 8);
    a.q[1] = *reinterpret_cast<const uint4*>(arow + k0 + 16 + hi * 8);
    b.q[0] = *reinterpret_cast<const uint4*>(brow + k0 + hi * 8);
    b.q[1] = *reinterpret_cast<const uint4*>(brow + k0 + 16 + hi * 8);
    acc = wmma16(a.v, b.v, acc);
  }
  float bn = bo[n0 + lq];
#pragma unroll
  for (int r = 0; r < 8; ++r) {
    int g = m0 + r + 8 * hi;
    size_t idx = (size_t)g * Dc + n0 + lq;
    xres[idx] = acc[r] + bn + tgt[idx];
  }
}

// ---------------------------------------------------------------------------
// ln: LayerNorm over D=256, one block per row
// ---------------------------------------------------------------------------
__global__ void __launch_bounds__(256)
ln_kernel(const float* __restrict__ xres, const float* __restrict__ gamma,
          const float* __restrict__ beta, float* __restrict__ out) {
  __shared__ float red[Dc];
  int row = blockIdx.x, tid = threadIdx.x;
  float x = xres[(size_t)row * Dc + tid];
  red[tid] = x;
  __syncthreads();
  for (int s = Dc / 2; s > 0; s >>= 1) {
    if (tid < s) red[tid] += red[tid + s];
    __syncthreads();
  }
  float mu = red[0] * (1.f / Dc);
  __syncthreads();
  float d = x - mu;
  red[tid] = d * d;
  __syncthreads();
  for (int s = Dc / 2; s > 0; s >>= 1) {
    if (tid < s) red[tid] += red[tid + s];
    __syncthreads();
  }
  float var = red[0] * (1.f / Dc);
  out[(size_t)row * Dc + tid] = d * rsqrtf(var + 1e-5f) * gamma[tid] + beta[tid];
}

// ---------------------------------------------------------------------------
extern "C" void kernel_launch(void* const* d_in, const int* in_sizes, int n_in,
                              void* d_out, int out_size, void* d_ws, size_t ws_size,
                              hipStream_t stream) {
  (void)in_sizes; (void)n_in; (void)out_size; (void)ws_size;
  const float* tgt   = (const float*)d_in[0];
  const float* qpos  = (const float*)d_in[1];
  const float* plocs = (const float*)d_in[2];
  const unsigned char* kpm = (const unsigned char*)d_in[3];
  const float* Wq = (const float*)d_in[4];  const float* bq = (const float*)d_in[5];
  const float* Wk = (const float*)d_in[6];  const float* bk = (const float*)d_in[7];
  const float* Wv = (const float*)d_in[8];  const float* bv = (const float*)d_in[9];
  const float* Wo = (const float*)d_in[10]; const float* bo = (const float*)d_in[11];
  const float* Wl = (const float*)d_in[12]; const float* bl = (const float*)d_in[13];
  const float* gamma = (const float*)d_in[14];
  const float* beta  = (const float*)d_in[15];
  float* out = (float*)d_out;

  char* ws = (char*)d_ws;
  const size_t MB = 1024u * 1024u;
  _Float16* tgtf  = (_Float16*)(ws + 0);                    // 4 MB
  _Float16* qkinf = (_Float16*)(ws + 4 * MB);               // 4 MB
  _Float16* Wt    = (_Float16*)(ws + 8 * MB);               // 512 KB (4 x 64K f16)
  _Float16* Qf    = (_Float16*)(ws + 8 * MB + 512 * 1024);  // 4 MB
  _Float16* Kf    = (_Float16*)(ws + 12 * MB + 512 * 1024); // 4 MB
  _Float16* VtT   = (_Float16*)(ws + 16 * MB + 512 * 1024); // 4 MB
  _Float16* attb  = (_Float16*)(ws + 20 * MB + 512 * 1024); // 4 MB
  float*    xres  = (float*)(ws + 0);  // 8 MB, reuses tgtf/qkinf (dead by oproj)

  prep_kernel<<<8192, 256, 0, stream>>>(tgt, qpos, Wq, Wk, Wv, Wo, tgtf, qkinf, Wt);
  proj_kernel<<<1024, 256, 0, stream>>>(qkinf, Wt + 0 * 65536, bq, Qf, nullptr, 0);
  proj_kernel<<<1024, 256, 0, stream>>>(qkinf, Wt + 1 * 65536, bk, Kf, nullptr, 0);
  proj_kernel<<<1024, 256, 0, stream>>>(tgtf,  Wt + 2 * 65536, bv, nullptr, VtT, 1);
  attn_kernel<<<512, 256, 0, stream>>>(Qf, Kf, VtT, plocs, kpm, Wl, bl, attb);
  oproj_kernel<<<1024, 256, 0, stream>>>(attb, Wt + 3 * 65536, bo, tgt, xres);
  ln_kernel<<<8192, 256, 0, stream>>>(xres, gamma, beta, out);
}
